// Bert_77988016160894
// MI455X (gfx1250) — matrix operand order repaired and checked by reading, not compiled, not run on hardware
//
#include <hip/hip_runtime.h>
#include <hip/hip_bf16.h>

// ---------------------------------------------------------------------------
// Model constants (match reference)
// ---------------------------------------------------------------------------
#define S_    4096
#define D_    768
#define H_    12
#define HD_   64
#define L_    2
#define C_    256
#define WIN_  256
#define NCLS_ 64
#define NC_   16
#define FF_   3072
#define NEGV  (-1.0e9f)
#define SCALE_ 0.125f   // 1/sqrt(64)

// ---------------------------------------------------------------------------
// Types / WMMA helpers (CDNA5 wave32; v_wmma_f32_16x16x32_bf16)
// ---------------------------------------------------------------------------
typedef __attribute__((ext_vector_type(16))) __bf16          v16bf;
typedef __attribute__((ext_vector_type(8)))  float           v8f;
typedef __attribute__((ext_vector_type(8)))  unsigned short  ushort8;
typedef __attribute__((ext_vector_type(4)))  unsigned short  us4;
typedef __attribute__((ext_vector_type(4)))  unsigned int    u32x4;
typedef __attribute__((ext_vector_type(8)))  int             i32x8;
typedef __attribute__((ext_vector_type(4)))  int             i32x4;

#ifndef __has_builtin
#define __has_builtin(x) 0
#endif
#if __has_builtin(__builtin_amdgcn_tensor_load_to_lds) && \
    __has_builtin(__builtin_amdgcn_s_wait_tensorcnt)
#define USE_TDM 1
#else
#define USE_TDM 0
#endif

union Frag16 { v16bf v; ushort8 h[2]; };

__device__ __forceinline__ unsigned short f2bf(float f) {
  union { float f; unsigned int u; } x; x.f = f;
  unsigned int u = x.u;
  return (unsigned short)((u + 0x7FFFu + ((u >> 16) & 1u)) >> 16);
}

// A fragment: 16(M)x32(K) bf16 from row-major [16][stride] LDS tile.
__device__ __forceinline__ v16bf frag_a_ld(const unsigned short* t, int stride, int lane) {
  int m  = lane & 15;
  int k0 = (lane >> 4) << 3;          // 0 or 8
  const unsigned short* p = t + m * stride + k0;
  Frag16 f;
  f.h[0] = *reinterpret_cast<const ushort8*>(p);
  f.h[1] = *reinterpret_cast<const ushort8*>(p + 16);
  return f.v;
}

// B fragment: 32(K)x16(N) bf16 from N-major [16][stride] LDS tile.
__device__ __forceinline__ v16bf frag_b_ld(const unsigned short* t, int stride, int lane) {
  int n  = lane & 15;
  int k0 = (lane >> 4) << 3;          // 0 or 8
  const unsigned short* p = t + n * stride + k0;
  Frag16 f;
  f.h[0] = *reinterpret_cast<const ushort8*>(p);
  f.h[1] = *reinterpret_cast<const ushort8*>(p + 16);
  return f.v;
}

__device__ __forceinline__ v8f wmma_bf16(v16bf a, v16bf b, v8f c) {
  return __builtin_amdgcn_wmma_f32_16x16x32_bf16(
      false, a, false, b, (short)0, c, false, false);
}

__device__ __forceinline__ float gelu_approx(float x) {
  float x3 = x * x * x;
  return 0.5f * x * (1.0f + tanhf(0.7978845608028654f * (x + 0.044715f * x3)));
}

#if USE_TDM
// ---------------------------------------------------------------------------
// Tensor Data Mover: 2D bf16 tile (tile_d0 contiguous elems x tile_d1 rows)
// global -> LDS. D# built per CDNA5 ISA (groups 0/1).
// data_size=1 (2 bytes). OOB rows/cols read as zero via tensor_dim clamps.
// ---------------------------------------------------------------------------
__device__ __forceinline__ void tdm_load_bf16_tile(
    const unsigned short* gsrc, unsigned lds_byte_off,
    unsigned long long row_stride_elems,
    unsigned dim0_rem, unsigned dim1_rem,
    unsigned tile_d0, unsigned tile_d1) {
  unsigned long long ga = (unsigned long long)(const void*)gsrc;
  u32x4 g0;
  g0[0] = 1u;                                            // count=1, user mode
  g0[1] = lds_byte_off;                                  // lds_addr
  g0[2] = (unsigned)(ga & 0xFFFFFFFFu);                  // global_addr[31:0]
  g0[3] = (unsigned)((ga >> 32) & 0x1FFFFFFu) | (2u << 30); // addr[56:32] | type=2
  i32x8 g1;
  g1[0] = (int)(1u << 16);                               // data_size=1 (2B)
  g1[1] = (int)((dim0_rem & 0xFFFFu) << 16);             // tensor_dim0[15:0]
  g1[2] = (int)(((dim0_rem >> 16) & 0xFFFFu) |
                ((dim1_rem & 0xFFFFu) << 16));           // dim0[31:16] | dim1[15:0]
  g1[3] = (int)(((dim1_rem >> 16) & 0xFFFFu) |
                ((tile_d0 & 0xFFFFu) << 16));            // dim1[31:16] | tile_dim0
  g1[4] = (int)(tile_d1 & 0xFFFFu);                      // tile_dim1 (tile_dim2=0)
  g1[5] = (int)(unsigned)(row_stride_elems & 0xFFFFFFFFull);  // dim0_stride[31:0]
  g1[6] = (int)(unsigned)((row_stride_elems >> 32) & 0xFFFFull);
  g1[7] = 0;
  i32x4 z4 = {0, 0, 0, 0};
#if __clang_major__ >= 23
  i32x8 z8 = {0, 0, 0, 0, 0, 0, 0, 0};
  __builtin_amdgcn_tensor_load_to_lds(g0, g1, z4, z4, z8, 0);
#else
  __builtin_amdgcn_tensor_load_to_lds(g0, g1, z4, z4, 0);
#endif
}
__device__ __forceinline__ unsigned lds_off_of(const void* p) {
  return (unsigned)(unsigned long long)p;
}
#endif

// ---------------------------------------------------------------------------
// Weight convert+transpose: W[K][N] f32 -> WT[N][K] bf16 (once per launch)
// ---------------------------------------------------------------------------
__global__ __launch_bounds__(256)
void wconv_t(const float* __restrict__ W, unsigned short* __restrict__ WT,
             int K, int N) {
  int i = blockIdx.x * 256 + threadIdx.x;
  if (i < K * N) {
    int kk = i / N, n = i - kk * N;
    WT[(size_t)n * K + kk] = f2bf(W[i]);
  }
}

// ---------------------------------------------------------------------------
// Embedding + LN : h(f32) and hbf(bf16 mirror)
// ---------------------------------------------------------------------------
__global__ __launch_bounds__(256)
void embed_ln(const int* __restrict__ x, const int* __restrict__ segs,
              const float* __restrict__ we, const float* __restrict__ pe,
              const float* __restrict__ te, const float* __restrict__ ls,
              const float* __restrict__ lb, float* __restrict__ h,
              unsigned short* __restrict__ hbf) {
  int s = blockIdx.x;
  int tok = x[s], seg = segs[s];
  __shared__ float red[256];
  float vals[3];
  float sum = 0.f;
#pragma unroll
  for (int i = 0; i < 3; ++i) {
    int d = threadIdx.x + i * 256;
    float v = we[(size_t)tok * D_ + d] + pe[(size_t)s * D_ + d] + te[(size_t)seg * D_ + d];
    vals[i] = v; sum += v;
  }
  red[threadIdx.x] = sum; __syncthreads();
  for (int o = 128; o > 0; o >>= 1) {
    if (threadIdx.x < o) red[threadIdx.x] += red[threadIdx.x + o];
    __syncthreads();
  }
  float mean = red[0] * (1.0f / D_);
  __syncthreads();
  float vs = 0.f;
#pragma unroll
  for (int i = 0; i < 3; ++i) { float d2 = vals[i] - mean; vs += d2 * d2; }
  red[threadIdx.x] = vs; __syncthreads();
  for (int o = 128; o > 0; o >>= 1) {
    if (threadIdx.x < o) red[threadIdx.x] += red[threadIdx.x + o];
    __syncthreads();
  }
  float rstd = rsqrtf(red[0] * (1.0f / D_) + 1e-5f);
#pragma unroll
  for (int i = 0; i < 3; ++i) {
    int d = threadIdx.x + i * 256;
    float o2 = (vals[i] - mean) * rstd * ls[d] + lb[d];
    h[(size_t)s * D_ + d] = o2;
    hbf[(size_t)s * D_ + d] = f2bf(o2);
  }
}

// ---------------------------------------------------------------------------
// dst(f32) = LN(x + y); also bf16 mirror
// ---------------------------------------------------------------------------
__global__ __launch_bounds__(256)
void add_ln(const float* __restrict__ x, const float* __restrict__ y,
            const float* __restrict__ s, const float* __restrict__ b,
            float* __restrict__ dst, unsigned short* __restrict__ dbf) {
  int t = blockIdx.x;
  __shared__ float red[256];
  float vals[3];
  float sum = 0.f;
#pragma unroll
  for (int i = 0; i < 3; ++i) {
    int d = threadIdx.x + i * 256;
    float v = x[(size_t)t * D_ + d] + y[(size_t)t * D_ + d];
    vals[i] = v; sum += v;
  }
  red[threadIdx.x] = sum; __syncthreads();
  for (int o = 128; o > 0; o >>= 1) {
    if (threadIdx.x < o) red[threadIdx.x] += red[threadIdx.x + o];
    __syncthreads();
  }
  float mean = red[0] * (1.0f / D_);
  __syncthreads();
  float vs = 0.f;
#pragma unroll
  for (int i = 0; i < 3; ++i) { float d2 = vals[i] - mean; vs += d2 * d2; }
  red[threadIdx.x] = vs; __syncthreads();
  for (int o = 128; o > 0; o >>= 1) {
    if (threadIdx.x < o) red[threadIdx.x] += red[threadIdx.x + o];
    __syncthreads();
  }
  float rstd = rsqrtf(red[0] * (1.0f / D_) + 1e-5f);
#pragma unroll
  for (int i = 0; i < 3; ++i) {
    int d = threadIdx.x + i * 256;
    float o2 = (vals[i] - mean) * rstd * s[d] + b[d];
    dst[(size_t)t * D_ + d] = o2;
    dbf[(size_t)t * D_ + d] = f2bf(o2);
  }
}

// ---------------------------------------------------------------------------
// GEMM: out[M,N] = act((Abf[M,K] @ WT[N,K]^T + bias) * scale)
// All-bf16 operands; TDM double-buffered tile pipeline (A:128x64, B:64x64).
// 128 threads = 4 waves; wave owns 32 rows; 16 WMMAs per 64-K step.
// ---------------------------------------------------------------------------
__global__ __launch_bounds__(128)
void gemm_wmma(const unsigned short* __restrict__ A,
               const unsigned short* __restrict__ WT,
               const float* __restrict__ bias, void* __restrict__ outp,
               int M, int N, int K, float out_scale, int act, int outbf) {
  __shared__ __align__(16) unsigned short As[2][128 * 64];  // 2x16KB
  __shared__ __align__(16) unsigned short Bs[2][64 * 64];   // 2x8KB, N-major
  const int tid = threadIdx.x, lane = tid & 31, w = tid >> 5;
  const int row0 = blockIdx.y * 128, col0 = blockIdx.x * 64;
  v8f acc[2][4] = {};
  const int KT = K / 64;
  int cur = 0;

#if USE_TDM
  unsigned aoff[2] = { lds_off_of(&As[0][0]), lds_off_of(&As[1][0]) };
  unsigned boff[2] = { lds_off_of(&Bs[0][0]), lds_off_of(&Bs[1][0]) };
  if (w == 0) {
    tdm_load_bf16_tile(A  + (size_t)row0 * K, aoff[0], (unsigned long long)K,
                       (unsigned)K, (unsigned)(M - row0), 64u, 128u);
    tdm_load_bf16_tile(WT + (size_t)col0 * K, boff[0], (unsigned long long)K,
                       (unsigned)K, (unsigned)(N - col0), 64u, 64u);
  }
#endif

  for (int t = 0; t < KT; ++t) {
    const int k0 = t * 64;
#if USE_TDM
    if (w == 0) {
      if (t + 1 < KT) {
        tdm_load_bf16_tile(A  + (size_t)row0 * K + k0 + 64, aoff[cur ^ 1],
                           (unsigned long long)K, (unsigned)(K - k0 - 64),
                           (unsigned)(M - row0), 64u, 128u);
        tdm_load_bf16_tile(WT + (size_t)col0 * K + k0 + 64, boff[cur ^ 1],
                           (unsigned long long)K, (unsigned)(K - k0 - 64),
                           (unsigned)(N - col0), 64u, 64u);
        __builtin_amdgcn_s_wait_tensorcnt(2);   // current tiles complete
      } else {
        __builtin_amdgcn_s_wait_tensorcnt(0);
      }
    }
#else
    for (int i = tid; i < 128 * 16; i += 128) {         // A: 128x64 shorts
      int r = i >> 4, c4 = (i & 15) << 2;
      int gr = row0 + r; gr = gr < M ? gr : M - 1;
      *(us4*)&As[cur][r * 64 + c4] = *(const us4*)&A[(size_t)gr * K + k0 + c4];
    }
    for (int i = tid; i < 64 * 16; i += 128) {          // B: 64x64 shorts
      int n = i >> 4, c4 = (i & 15) << 2;
      *(us4*)&Bs[cur][n * 64 + c4] = *(const us4*)&WT[(size_t)(col0 + n) * K + k0 + c4];
    }
#endif
    __syncthreads();
    const unsigned short* At = &As[cur][(w * 32) * 64];
#pragma unroll
    for (int kk = 0; kk < 64; kk += 32) {
      v16bf a0 = frag_a_ld(At + kk, 64, lane);
      v16bf a1 = frag_a_ld(At + 16 * 64 + kk, 64, lane);
#pragma unroll
      for (int nn = 0; nn < 4; ++nn) {
        v16bf b = frag_b_ld(&Bs[cur][nn * 16 * 64 + kk], 64, lane);
        acc[0][nn] = wmma_bf16(a0, b, acc[0][nn]);
        acc[1][nn] = wmma_bf16(a1, b, acc[1][nn]);
      }
    }
    __syncthreads();
    cur ^= 1;
  }

  const int mo = (lane >> 4) << 3;
  const int nc = lane & 15;
#pragma unroll
  for (int mt = 0; mt < 2; ++mt)
#pragma unroll
    for (int nn = 0; nn < 4; ++nn) {
      int col = col0 + nn * 16 + nc;
      float bb = bias[col];
#pragma unroll
      for (int vr = 0; vr < 8; ++vr) {
        int row = row0 + w * 32 + mt * 16 + mo + vr;
        if (row < M) {
          float v = (acc[mt][nn][vr] + bb) * out_scale;
          if (act) v = gelu_approx(v);
          if (outbf) ((unsigned short*)outp)[(size_t)row * N + col] = f2bf(v);
          else       ((float*)outp)[(size_t)row * N + col] = v;
        }
      }
    }
}

// ---------------------------------------------------------------------------
// Band attention (global-key block + banded keys), flash style, bf16 in/out.
// Grid (NC, H); 512 threads = 16 waves; wave = one 16-query tile.
// Key space: [64 global | 768 band] = 832 = 26 steps of 32.
// ---------------------------------------------------------------------------
__global__ __launch_bounds__(512)
void band_attn(const unsigned short* __restrict__ q, const unsigned short* __restrict__ k,
               const unsigned short* __restrict__ v, const int* __restrict__ gidx,
               const int* __restrict__ mask, const int* __restrict__ iglb,
               unsigned short* __restrict__ out) {
  const int chunk = blockIdx.x;
  const int hh = blockIdx.y;
  const int tid = threadIdx.x;
  const int lane = tid & 31, w = tid >> 5;
  const int hoff = hh * HD_;
  __shared__ __align__(16) unsigned short Qs[C_ * HD_];     // 32KB
  __shared__ __align__(16) unsigned short Ks[32 * HD_];     // [key][hd]
  __shared__ __align__(16) unsigned short Vs[HD_ * 32];     // [hd][key]
  __shared__ __align__(16) unsigned short Ps[16 * 16 * 32]; // per-wave P tiles
  __shared__ unsigned char Kok[32];

  for (int i = tid; i < C_ * 16; i += 512) {
    int r = i >> 4, c4 = (i & 15) << 2;
    *(us4*)&Qs[r * 64 + c4] = *(const us4*)&q[(size_t)(chunk * C_ + r) * D_ + hoff + c4];
  }
  __syncthreads();

  float Mx[8], Lx[8];
  v8f o[4] = {};
#pragma unroll
  for (int i = 0; i < 8; ++i) { Mx[i] = -1e30f; Lx[i] = 0.f; }
  const int mo = (lane >> 4) << 3;
  const int nc = lane & 15;
  const int n0 = chunk * C_ - C_;
  const int NSTEPS = (NCLS_ + 3 * C_) / 32;   // 26

  for (int st = 0; st < NSTEPS; ++st) {
    const int colB = st * 32;
    {   // 512 threads load 32 keys x 64 hd (one us4 each)
      int key = tid >> 4, c4 = (tid & 15) << 2;
      int col = colB + key;
      int src, kv;
      if (col < NCLS_) { src = gidx[col]; kv = (mask[src] > 0); }
      else {
        int jj = col - NCLS_; src = n0 + jj;
        kv = (src >= 0 && src < S_) ? ((mask[src] > 0) && (iglb[src] == 0)) : 0;
      }
      us4 kf = {0, 0, 0, 0}, vf = {0, 0, 0, 0};
      if (kv) {
        kf = *(const us4*)&k[(size_t)src * D_ + hoff + c4];
        vf = *(const us4*)&v[(size_t)src * D_ + hoff + c4];
      }
      *(us4*)&Ks[key * 64 + c4] = kf;
      Vs[(c4 + 0) * 32 + key] = vf[0];
      Vs[(c4 + 1) * 32 + key] = vf[1];
      Vs[(c4 + 2) * 32 + key] = vf[2];
      Vs[(c4 + 3) * 32 + key] = vf[3];
      if (c4 == 0) Kok[key] = (unsigned char)kv;
    }
    __syncthreads();

    const unsigned short* Qt = Qs + w * 16 * HD_;
    v16bf a0 = frag_a_ld(Qt, HD_, lane);
    v16bf a1 = frag_a_ld(Qt + 32, HD_, lane);
    v8f s0 = {}, s1 = {};
    s0 = wmma_bf16(a0, frag_b_ld(Ks, HD_, lane), s0);
    s0 = wmma_bf16(a1, frag_b_ld(Ks + 32, HD_, lane), s0);
    s1 = wmma_bf16(a0, frag_b_ld(Ks + 16 * HD_, HD_, lane), s1);
    s1 = wmma_bf16(a1, frag_b_ld(Ks + 16 * HD_ + 32, HD_, lane), s1);

    float p0[8], p1[8], alpha[8];
    int c0 = colB + nc, c1 = c0 + 16;
    int kok0 = Kok[nc], kok1 = Kok[nc + 16];
#pragma unroll
    for (int vr = 0; vr < 8; ++vr) {
      int qi = w * 16 + mo + vr;
      int ok0 = kok0, ok1 = kok1;
      if (c0 >= NCLS_) { int rel = (c0 - NCLS_) - (qi + C_); ok0 = ok0 && (rel <= WIN_) && (rel >= -WIN_); }
      if (c1 >= NCLS_) { int rel = (c1 - NCLS_) - (qi + C_); ok1 = ok1 && (rel <= WIN_) && (rel >= -WIN_); }
      float x0 = ok0 ? s0[vr] : NEGV;
      float x1 = ok1 ? s1[vr] : NEGV;
      float tm = fmaxf(x0, x1);
      for (int off = 1; off < 16; off <<= 1) tm = fmaxf(tm, __shfl_xor(tm, off, 32));
      float nM = fmaxf(Mx[vr], tm);
      float al = __expf(Mx[vr] - nM);
      float e0 = __expf(x0 - nM), e1 = __expf(x1 - nM);
      float rs = e0 + e1;
      for (int off = 1; off < 16; off <<= 1) rs += __shfl_xor(rs, off, 32);
      Mx[vr] = nM;
      Lx[vr] = Lx[vr] * al + rs;
      alpha[vr] = al; p0[vr] = e0; p1[vr] = e1;
    }
#pragma unroll
    for (int nn = 0; nn < 4; ++nn)
#pragma unroll
      for (int vr = 0; vr < 8; ++vr) o[nn][vr] *= alpha[vr];

    unsigned short* Pw = Ps + w * 16 * 32;
#pragma unroll
    for (int vr = 0; vr < 8; ++vr) {
      int m = mo + vr;
      Pw[m * 32 + nc]      = f2bf(p0[vr]);
      Pw[m * 32 + nc + 16] = f2bf(p1[vr]);
    }
    v16bf pa = frag_a_ld(Pw, 32, lane);
#pragma unroll
    for (int nn = 0; nn < 4; ++nn)
      o[nn] = wmma_bf16(pa, frag_b_ld(Vs + nn * 16 * 32, 32, lane), o[nn]);
    __syncthreads();
  }

#pragma unroll
  for (int nn = 0; nn < 4; ++nn)
#pragma unroll
    for (int vr = 0; vr < 8; ++vr) {
      int tokrow = chunk * C_ + w * 16 + mo + vr;
      out[(size_t)tokrow * D_ + hoff + nn * 16 + nc] = f2bf(o[nn][vr] / Lx[vr]);
    }
}

// ---------------------------------------------------------------------------
// Global-token full attention (64 queries x 4096 keys), bf16 in/out.
// Grid H; 128 threads = 4 waves; wave = one 16-query tile.
// ---------------------------------------------------------------------------
__global__ __launch_bounds__(128)
void glb_attn(const unsigned short* __restrict__ qg, const unsigned short* __restrict__ kg,
              const unsigned short* __restrict__ vg, const int* __restrict__ gidx,
              const int* __restrict__ mask, unsigned short* __restrict__ out) {
  const int hh = blockIdx.x;
  const int tid = threadIdx.x;
  const int lane = tid & 31, w = tid >> 5;
  const int hoff = hh * HD_;
  __shared__ __align__(16) unsigned short Qs[NCLS_ * HD_];
  __shared__ __align__(16) unsigned short Ks[32 * HD_];
  __shared__ __align__(16) unsigned short Vs[HD_ * 32];
  __shared__ __align__(16) unsigned short Ps[4 * 16 * 32];
  __shared__ unsigned char Kok[32];

  for (int i = tid; i < NCLS_ * 16; i += 128) {
    int r = i >> 4, c4 = (i & 15) << 2;
    *(us4*)&Qs[r * 64 + c4] = *(const us4*)&qg[(size_t)r * D_ + hoff + c4];
  }
  __syncthreads();

  float Mx[8], Lx[8];
  v8f o[4] = {};
#pragma unroll
  for (int i = 0; i < 8; ++i) { Mx[i] = -1e30f; Lx[i] = 0.f; }
  const int mo = (lane >> 4) << 3;
  const int nc = lane & 15;

  for (int st = 0; st < S_ / 32; ++st) {
    const int colB = st * 32;
    for (int i = tid; i < 32 * 16; i += 128) {
      int key = i >> 4, c4 = (i & 15) << 2;
      int src = colB + key;
      int kv = (mask[src] > 0);
      us4 kf = {0, 0, 0, 0}, vf = {0, 0, 0, 0};
      if (kv) {
        kf = *(const us4*)&kg[(size_t)src * D_ + hoff + c4];
        vf = *(const us4*)&vg[(size_t)src * D_ + hoff + c4];
      }
      *(us4*)&Ks[key * 64 + c4] = kf;
      Vs[(c4 + 0) * 32 + key] = vf[0];
      Vs[(c4 + 1) * 32 + key] = vf[1];
      Vs[(c4 + 2) * 32 + key] = vf[2];
      Vs[(c4 + 3) * 32 + key] = vf[3];
      if (c4 == 0) Kok[key] = (unsigned char)kv;
    }
    __syncthreads();

    const unsigned short* Qt = Qs + w * 16 * HD_;
    v16bf a0 = frag_a_ld(Qt, HD_, lane);
    v16bf a1 = frag_a_ld(Qt + 32, HD_, lane);
    v8f s0 = {}, s1 = {};
    s0 = wmma_bf16(a0, frag_b_ld(Ks, HD_, lane), s0);
    s0 = wmma_bf16(a1, frag_b_ld(Ks + 32, HD_, lane), s0);
    s1 = wmma_bf16(a0, frag_b_ld(Ks + 16 * HD_, HD_, lane), s1);
    s1 = wmma_bf16(a1, frag_b_ld(Ks + 16 * HD_ + 32, HD_, lane), s1);

    float p0[8], p1[8], alpha[8];
    int kok0 = Kok[nc], kok1 = Kok[nc + 16];
#pragma unroll
    for (int vr = 0; vr < 8; ++vr) {
      float x0 = kok0 ? s0[vr] : NEGV;
      float x1 = kok1 ? s1[vr] : NEGV;
      float tm = fmaxf(x0, x1);
      for (int off = 1; off < 16; off <<= 1) tm = fmaxf(tm, __shfl_xor(tm, off, 32));
      float nM = fmaxf(Mx[vr], tm);
      float al = __expf(Mx[vr] - nM);
      float e0 = __expf(x0 - nM), e1 = __expf(x1 - nM);
      float rs = e0 + e1;
      for (int off = 1; off < 16; off <<= 1) rs += __shfl_xor(rs, off, 32);
      Mx[vr] = nM; Lx[vr] = Lx[vr] * al + rs;
      alpha[vr] = al; p0[vr] = e0; p1[vr] = e1;
    }
#pragma unroll
    for (int nn = 0; nn < 4; ++nn)
#pragma unroll
      for (int vr = 0; vr < 8; ++vr) o[nn][vr] *= alpha[vr];

    unsigned short* Pw = Ps + w * 16 * 32;
#pragma unroll
    for (int vr = 0; vr < 8; ++vr) {
      int m = mo + vr;
      Pw[m * 32 + nc]      = f2bf(p0[vr]);
      Pw[m * 32 + nc + 16] = f2bf(p1[vr]);
    }
    v16bf pa = frag_a_ld(Pw, 32, lane);
#pragma unroll
    for (int nn = 0; nn < 4; ++nn)
      o[nn] = wmma_bf16(pa, frag_b_ld(Vs + nn * 16 * 32, 32, lane), o[nn]);
    __syncthreads();
  }

#pragma unroll
  for (int nn = 0; nn < 4; ++nn)
#pragma unroll
    for (int vr = 0; vr < 8; ++vr) {
      int qrow = w * 16 + mo + vr;
      int tok = gidx[qrow];
      out[(size_t)tok * D_ + hoff + nn * 16 + nc] = f2bf(o[nn][vr] / Lx[vr]);
    }
}

// ---------------------------------------------------------------------------
// Small helpers
// ---------------------------------------------------------------------------
__global__ void gather_rows(const unsigned short* __restrict__ hbf,
                            const int* __restrict__ gidx,
                            unsigned short* __restrict__ out) {
  int i = blockIdx.x * blockDim.x + threadIdx.x;
  if (i < NCLS_ * D_) {
    int r = i / D_, c = i - r * D_;
    out[i] = hbf[(size_t)gidx[r] * D_ + c];
  }
}
__global__ void iglb_init(int* g) {
  int i = blockIdx.x * blockDim.x + threadIdx.x;
  if (i < S_) g[i] = 0;
}
__global__ void iglb_set(int* g, const int* __restrict__ clss) {
  int i = threadIdx.x;
  if (i < NCLS_) g[clss[i]] = 1;
}

// ---------------------------------------------------------------------------
// Orchestration
// ---------------------------------------------------------------------------
extern "C" void kernel_launch(void* const* d_in, const int* in_sizes, int n_in,
                              void* d_out, int out_size, void* d_ws, size_t ws_size,
                              hipStream_t stream) {
  const int*   x     = (const int*)d_in[0];
  const int*   mask  = (const int*)d_in[1];
  const int*   clss  = (const int*)d_in[2];
  const int*   segs  = (const int*)d_in[3];
  const float* we    = (const float*)d_in[4];
  const float* pe    = (const float*)d_in[5];
  const float* te    = (const float*)d_in[6];
  const float* lnes  = (const float*)d_in[7];
  const float* lneb  = (const float*)d_in[8];
  const float* Wq    = (const float*)d_in[9];
  const float* bq    = (const float*)d_in[10];
  const float* Wk    = (const float*)d_in[11];
  const float* bk    = (const float*)d_in[12];
  const float* Wv    = (const float*)d_in[13];
  const float* bv    = (const float*)d_in[14];
  const float* Wqg   = (const float*)d_in[15];
  const float* bqg   = (const float*)d_in[16];
  const float* Wkg   = (const float*)d_in[17];
  const float* bkg   = (const float*)d_in[18];
  const float* Wvg   = (const float*)d_in[19];
  const float* bvg   = (const float*)d_in[20];
  const float* Wo    = (const float*)d_in[21];
  const float* bo    = (const float*)d_in[22];
  const float* ln1s  = (const float*)d_in[23];
  const float* ln1b  = (const float*)d_in[24];
  const float* Wf1   = (const float*)d_in[25];
  const float* bf1   = (const float*)d_in[26];
  const float* Wf2   = (const float*)d_in[27];
  const float* bf2   = (const float*)d_in[28];
  const float* ln2s  = (const float*)d_in[29];
  const float* ln2b  = (const float*)d_in[30];

  const size_t SD = (size_t)S_ * D_;
  const size_t DD = (size_t)D_ * D_;
  const size_t DF = (size_t)D_ * FF_;

  // workspace layout (256B aligned slices)
  char* base = (char*)d_ws;
  size_t off = 0;
  auto alloc = [&](size_t bytes) {
    char* p = base + off;
    off = (off + bytes + 255) & ~(size_t)255;
    return p;
  };
  float*          h     = (float*)alloc(SD * 4);
  float*          proj  = (float*)alloc(SD * 4);
  unsigned short* hbf   = (unsigned short*)alloc(SD * 2);
  unsigned short* qb    = (unsigned short*)alloc(SD * 2);   // q  -> kgf
  unsigned short* kb    = (unsigned short*)alloc(SD * 2);   // k  -> vgf
  unsigned short* vb    = (unsigned short*)alloc(SD * 2);
  unsigned short* attnb = (unsigned short*)alloc(SD * 2);
  unsigned short* ffb   = (unsigned short*)alloc((size_t)S_ * FF_ * 2);
  unsigned short* hg    = (unsigned short*)alloc((size_t)NCLS_ * D_ * 2);
  unsigned short* qgb   = (unsigned short*)alloc((size_t)NCLS_ * D_ * 2);
  int*            iglb  = (int*)alloc((size_t)S_ * 4);
  unsigned short* WT    = (unsigned short*)alloc((size_t)L_ * (7 * DD + 2 * DF) * 2);

  // -- one-time weight convert + transpose (bf16, [N][K]) --
  const size_t PL = 7 * DD + 2 * DF;      // per-layer WT elements
  const int gDD = (int)((DD + 255) / 256), gDF = (int)((DF + 255) / 256);
  for (int l = 0; l < L_; ++l) {
    unsigned short* wt = WT + (size_t)l * PL;
    wconv_t<<<gDD, 256, 0, stream>>>(Wq  + l * DD, wt + 0 * DD, D_, D_);
    wconv_t<<<gDD, 256, 0, stream>>>(Wk  + l * DD, wt + 1 * DD, D_, D_);
    wconv_t<<<gDD, 256, 0, stream>>>(Wv  + l * DD, wt + 2 * DD, D_, D_);
    wconv_t<<<gDD, 256, 0, stream>>>(Wqg + l * DD, wt + 3 * DD, D_, D_);
    wconv_t<<<gDD, 256, 0, stream>>>(Wkg + l * DD, wt + 4 * DD, D_, D_);
    wconv_t<<<gDD, 256, 0, stream>>>(Wvg + l * DD, wt + 5 * DD, D_, D_);
    wconv_t<<<gDD, 256, 0, stream>>>(Wo  + l * DD, wt + 6 * DD, D_, D_);
    wconv_t<<<gDF, 256, 0, stream>>>(Wf1 + l * DF, wt + 7 * DD, D_, FF_);
    wconv_t<<<gDF, 256, 0, stream>>>(Wf2 + l * DF, wt + 7 * DD + DF, FF_, D_);
  }

  embed_ln<<<S_, 256, 0, stream>>>(x, segs, we, pe, te, lnes, lneb, h, hbf);
  iglb_init<<<S_ / 256, 256, 0, stream>>>(iglb);
  iglb_set<<<1, NCLS_, 0, stream>>>(iglb, clss);

  const dim3 gP(D_ / 64, S_ / 128);     // SxD
  const dim3 gF1(FF_ / 64, S_ / 128);   // SxFF
  const dim3 gSm(D_ / 64, 1);           // NCLSxD

  for (int l = 0; l < L_; ++l) {
    const unsigned short* wt = WT + (size_t)l * PL;
    gemm_wmma<<<gP, 128, 0, stream>>>(hbf, wt + 0 * DD, bq + (size_t)l * D_, qb, S_, D_, D_, SCALE_, 0, 1);
    gemm_wmma<<<gP, 128, 0, stream>>>(hbf, wt + 1 * DD, bk + (size_t)l * D_, kb, S_, D_, D_, 1.f, 0, 1);
    gemm_wmma<<<gP, 128, 0, stream>>>(hbf, wt + 2 * DD, bv + (size_t)l * D_, vb, S_, D_, D_, 1.f, 0, 1);
    band_attn<<<dim3(NC_, H_), 512, 0, stream>>>(qb, kb, vb, clss, mask, iglb, attnb);

    gather_rows<<<(NCLS_ * D_ + 255) / 256, 256, 0, stream>>>(hbf, clss, hg);
    gemm_wmma<<<gSm, 128, 0, stream>>>(hg, wt + 3 * DD, bqg + (size_t)l * D_, qgb, NCLS_, D_, D_, SCALE_, 0, 1);
    gemm_wmma<<<gP, 128, 0, stream>>>(hbf, wt + 4 * DD, bkg + (size_t)l * D_, qb, S_, D_, D_, 1.f, 0, 1);
    gemm_wmma<<<gP, 128, 0, stream>>>(hbf, wt + 5 * DD, bvg + (size_t)l * D_, kb, S_, D_, D_, 1.f, 0, 1);
    glb_attn<<<H_, 128, 0, stream>>>(qgb, qb, kb, clss, mask, attnb);

    gemm_wmma<<<gP, 128, 0, stream>>>(attnb, wt + 6 * DD, bo + (size_t)l * D_, proj, S_, D_, D_, 1.f, 0, 0);
    add_ln<<<S_, 256, 0, stream>>>(h, proj, ln1s + (size_t)l * D_, ln1b + (size_t)l * D_, h, hbf);

    gemm_wmma<<<gF1, 128, 0, stream>>>(hbf, wt + 7 * DD, bf1 + (size_t)l * FF_, ffb, S_, FF_, D_, 1.f, 1, 1);
    gemm_wmma<<<gP, 128, 0, stream>>>(ffb, wt + 7 * DD + DF, bf2 + (size_t)l * D_, proj, S_, D_, FF_, 1.f, 0, 0);
    float* dst = (l == L_ - 1) ? (float*)d_out : h;
    add_ln<<<S_, 256, 0, stream>>>(h, proj, ln2s + (size_t)l * D_, ln2b + (size_t)l * D_, dst, hbf);
  }
}
